// Qwen3NextGatedDeltaNet_44255343018188
// MI455X (gfx1250) — matrix operand order repaired and checked by reading, not compiled, not run on hardware
//
#include <hip/hip_runtime.h>

// ---------------------------------------------------------------------------
// Types
// ---------------------------------------------------------------------------
typedef __attribute__((ext_vector_type(16))) __bf16          v16bf;
typedef __attribute__((ext_vector_type(8)))  float           v8f;
typedef __attribute__((ext_vector_type(8)))  unsigned short  ushort8;
typedef __attribute__((ext_vector_type(4)))  unsigned int    u32x4;
typedef __attribute__((ext_vector_type(8)))  int             i32x8;
typedef __attribute__((ext_vector_type(4)))  int             i32x4;

union FragU { v16bf v; ushort8 h[2]; };

// ---------------------------------------------------------------------------
// Scalar helpers
// ---------------------------------------------------------------------------
__device__ inline unsigned short f32_to_bf16bits(float f) {
  union { float f; unsigned int u; } v; v.f = f;
  unsigned int u = v.u;
  u += 0x7fffu + ((u >> 16) & 1u);           // round-to-nearest-even
  return (unsigned short)(u >> 16);
}
__device__ inline float bf16bits_to_f32(unsigned short h) {
  union { unsigned int u; float f; } v; v.u = ((unsigned int)h) << 16;
  return v.f;
}
__device__ inline float silu_f(float x) { return x / (1.f + __expf(-x)); }

// Fragment loaders (LDS, bf16). A-matrix 16x32 lane layout per ISA 7.12.2:
// lane<16 holds K {0..7,16..23} of row M=lane; lane>=16 holds K {8..15,24..31}.
// B-matrix 32x16: same K split as A, for column N=lane%16.
__device__ inline FragU ld_afrag(const unsigned short* base, int stride,
                                 int row, int kbase, int kg) {
  FragU f;
  f.h[0] = *(const ushort8*)(base + row * stride + kbase + kg * 8);
  f.h[1] = *(const ushort8*)(base + row * stride + kbase + 16 + kg * 8);
  return f;
}
__device__ inline FragU ld_bfrag(const unsigned short* base, int stride,
                                 int col, int kbase, int kg) {
  FragU f;
  f.h[0] = *(const ushort8*)(base + col * stride + kbase + kg * 8);
  f.h[1] = *(const ushort8*)(base + col * stride + kbase + 16 + kg * 8);
  return f;
}

#define WMMA_BF16(acc, fa, fb)                                               \
  (acc) = __builtin_amdgcn_wmma_f32_16x16x32_bf16(false, (fa).v, false,      \
                                                  (fb).v, (short)0, (acc),   \
                                                  false, false)

// ---------------------------------------------------------------------------
// Problem constants
// ---------------------------------------------------------------------------
#define NTOK   4096      // B*T
#define TSEQ   2048
#define DMODEL 2048
#define HK_    16
#define HV_    32
#define KD_    128
#define VD_    128
#define QKVZ_N 12288
#define QKCH   8192
#define CHUNK  64

// ---------------------------------------------------------------------------
// Generic WMMA GEMM:  C[M,N] = A[M,Kd] * B[Kd,N]
// ---------------------------------------------------------------------------
template<bool A_BF16, bool C_BF16>
__global__ __launch_bounds__(256)
void gemm_wmma(const void* __restrict__ Ap, const float* __restrict__ Bp,
               void* __restrict__ Cp, int M, int N, int Kd)
{
  __shared__ unsigned short As[128][40];   // [m][k], padded
  __shared__ unsigned short Bs[128][40];   // [n][k], padded (transposed tile)

  const int tid  = threadIdx.x;
  const int lane = tid & 31;
  const int wave = tid >> 5;
  const int wm   = wave & 3;
  const int wn   = wave >> 2;
  const int m0   = blockIdx.y * 128;
  const int n0   = blockIdx.x * 128;

  v8f acc[2][4];
  const v8f vzero = {0.f,0.f,0.f,0.f,0.f,0.f,0.f,0.f};
#pragma unroll
  for (int i = 0; i < 2; ++i)
#pragma unroll
    for (int j = 0; j < 4; ++j) acc[i][j] = vzero;

  const float*          Af = (const float*)Ap;
  const unsigned short* Ah = (const unsigned short*)Ap;

  for (int k0 = 0; k0 < Kd; k0 += 32) {
    if (k0 + 32 < Kd) {
      __builtin_prefetch((const char*)Bp + ((size_t)(k0 + 32) * N + n0) * 4, 0, 1);
      if (A_BF16)
        __builtin_prefetch((const char*)Ah + ((size_t)(m0 + (tid >> 1)) * Kd + k0 + 32) * 2, 0, 1);
      else
        __builtin_prefetch((const char*)Af + ((size_t)(m0 + (tid >> 1)) * Kd + k0 + 32) * 4, 0, 1);
    }

    if (A_BF16) {
#pragma unroll
      for (int j = 0; j < 2; ++j) {
        int f = j * 256 + tid;
        int e = f * 8;
        int r = e >> 5, c = e & 31;
        uint4 d = *(const uint4*)(Ah + (size_t)(m0 + r) * Kd + k0 + c);
        *(ushort8*)&As[r][c] = *(const ushort8*)&d;
      }
    } else {
#pragma unroll
      for (int j = 0; j < 4; ++j) {
        int f = j * 256 + tid;
        int e = f * 4;
        int r = e >> 5, c = e & 31;
        float4 d = *(const float4*)(Af + (size_t)(m0 + r) * Kd + k0 + c);
        As[r][c+0] = f32_to_bf16bits(d.x);
        As[r][c+1] = f32_to_bf16bits(d.y);
        As[r][c+2] = f32_to_bf16bits(d.z);
        As[r][c+3] = f32_to_bf16bits(d.w);
      }
    }

#pragma unroll
    for (int j = 0; j < 4; ++j) {
      int f  = j * 256 + tid;
      int kr = f >> 5;
      int c  = (f & 31) * 4;
      const size_t base = (size_t)(k0 + kr) * N + n0 + c;
#pragma unroll
      for (int u = 0; u < 4; ++u) {
        float x = (n0 + c + u < N) ? Bp[base + u] : 0.f;
        Bs[c + u][kr] = f32_to_bf16bits(x);
      }
    }
    __syncthreads();

    const int kg = lane >> 4;
    const int rm = lane & 15;
    FragU a[2], b[4];
#pragma unroll
    for (int t = 0; t < 2; ++t)
      a[t] = ld_afrag(&As[0][0], 40, wm * 32 + t * 16 + rm, 0, kg);
#pragma unroll
    for (int t2 = 0; t2 < 4; ++t2)
      b[t2] = ld_bfrag(&Bs[0][0], 40, wn * 64 + t2 * 16 + rm, 0, kg);
#pragma unroll
    for (int t = 0; t < 2; ++t)
#pragma unroll
      for (int t2 = 0; t2 < 4; ++t2)
        WMMA_BF16(acc[t][t2], a[t], b[t2]);
    __syncthreads();
  }

  const int rm = lane & 15;
  const int rh = (lane >> 4) * 8;
#pragma unroll
  for (int t = 0; t < 2; ++t)
#pragma unroll
    for (int t2 = 0; t2 < 4; ++t2) {
      int col = n0 + wn * 64 + t2 * 16 + rm;
      if (col < N) {
#pragma unroll
        for (int i = 0; i < 8; ++i) {
          int row = m0 + wm * 32 + t * 16 + rh + i;
          if (C_BF16)
            ((unsigned short*)Cp)[(size_t)row * N + col] = f32_to_bf16bits(acc[t][t2][i]);
          else
            ((float*)Cp)[(size_t)row * N + col] = acc[t][t2][i];
        }
      }
    }
}

// ---------------------------------------------------------------------------
// Fused causal conv(4) + SiLU + per-head l2norm + gates. One block per token.
// ---------------------------------------------------------------------------
__global__ __launch_bounds__(256)
void conv_gate_kernel(const unsigned short* __restrict__ qkvz,
                      const float* __restrict__ ba,
                      const float* __restrict__ conv_w,
                      const float* __restrict__ A_log,
                      const float* __restrict__ dt_bias,
                      unsigned short* __restrict__ qn,
                      unsigned short* __restrict__ kn,
                      unsigned short* __restrict__ vn,
                      float* __restrict__ gout,
                      float* __restrict__ bout)
{
  __shared__ float qk[4096];
  const int tok = blockIdx.x;
  const int t   = tok & (TSEQ - 1);
  const int tid = threadIdx.x;

  for (int c = tid; c < QKCH; c += 256) {
    float acc = 0.f;
#pragma unroll
    for (int j = 0; j < 4; ++j) {
      int tt = t - 3 + j;
      if (tt >= 0)
        acc += bf16bits_to_f32(qkvz[(size_t)(tok - 3 + j) * QKVZ_N + c]) * conv_w[c * 4 + j];
    }
    float s = silu_f(acc);
    if (c < 4096) qk[c] = s;
    else          vn[(size_t)tok * 4096 + (c - 4096)] = f32_to_bf16bits(s);
  }
  __syncthreads();

  const int lane = tid & 31, wave = tid >> 5;
#pragma unroll
  for (int hh = 0; hh < 2; ++hh) {
    const int h = wave * 2 + hh;
    float ss = 0.f;
    for (int i = lane; i < KD_; i += 32) { float x = qk[h * KD_ + i]; ss += x * x; }
#pragma unroll
    for (int m = 16; m >= 1; m >>= 1) ss += __shfl_xor(ss, m, 32);
    float rq = rsqrtf(ss + 1e-6f) * 0.08838834764831845f;   // * K^-1/2
    for (int i = lane; i < KD_; i += 32)
      qn[(size_t)tok * 2048 + h * KD_ + i] = f32_to_bf16bits(qk[h * KD_ + i] * rq);
    ss = 0.f;
    for (int i = lane; i < KD_; i += 32) { float x = qk[2048 + h * KD_ + i]; ss += x * x; }
#pragma unroll
    for (int m = 16; m >= 1; m >>= 1) ss += __shfl_xor(ss, m, 32);
    float rk = rsqrtf(ss + 1e-6f);
    for (int i = lane; i < KD_; i += 32)
      kn[(size_t)tok * 2048 + h * KD_ + i] = f32_to_bf16bits(qk[2048 + h * KD_ + i] * rk);
  }

  if (tid < HV_) {
    const int h = tid;
    float bv = ba[(size_t)tok * 64 + h];
    float av = ba[(size_t)tok * 64 + 32 + h] + dt_bias[h];
    bout[(size_t)tok * HV_ + h] = 1.f / (1.f + __expf(-bv));
    float sp = (av > 20.f) ? av : log1pf(__expf(av));
    gout[(size_t)tok * HV_ + h] = -__expf(A_log[h]) * sp;
  }
}

// ---------------------------------------------------------------------------
// Chunked gated delta rule (WY/UT transform), chunk C=64, WMMA throughout.
// One block per (b, value-head). State S^T [v][k] kept f32 + bf16 mirror in
// LDS (~259 KB of the 320 KB CDNA5 WGP LDS).
//
// Per chunk (gc = inclusive in-chunk cumsum of g, ec_i = exp(gc_i) <= 1):
//   A_ij   = -(k_i.k_j) exp(gc_i-gc_j)  (i>j, stored NEGATED, bf16)
//   attn_ij=  (q_i.k_j) exp(gc_i-gc_j)  (i>=j, bf16)
//   rhs    = V - ec o (K S0)                         [WMMA]
//   o1     = ec o (Q S0)                             [WMMA]
//   Delta  : blocked fwd substitution; cross-block corrections via WMMA
//            (future Delta rows pre-zeroed), 16x16 in-block solve is
//            lane-private per V-column (no barriers)
//   o      = o1 + attn . Delta                       [WMMA]
//   S^T    = gamma S^T + Delta^T . Kbar              [WMMA], gamma=ec_63
// ---------------------------------------------------------------------------
__global__ __launch_bounds__(256)
void delta_chunk_kernel(const unsigned short* __restrict__ qn,
                        const unsigned short* __restrict__ kn,
                        const unsigned short* __restrict__ vn,
                        const float* __restrict__ g,
                        const float* __restrict__ beta,
                        float* __restrict__ o)
{
  __shared__ float          Sf[128 * 128];     // S^T f32 master
  __shared__ unsigned short Sh[128 * 136];     // S^T bf16 operand mirror
  __shared__ unsigned short Qh[64 * 136];      // chunk Q (rows, k)
  __shared__ unsigned short Kh[64 * 136];      // chunk K
  __shared__ unsigned short UhT[128 * 72];     // Delta^T [v][j]
  __shared__ unsigned short Kt[128 * 72];      // Kbar^T [k][j] = er_j * K[j][k]
  __shared__ unsigned short Ah[64 * 72];       // -A strict-lower, bf16
  __shared__ unsigned short Th[64 * 72];       // attn lower, bf16
  __shared__ float Rf[64 * 128];               // rhs f32
  __shared__ float Yf[16 * 128];               // per-block substitution RHS
  __shared__ float Of[64 * 128];               // o1 f32
  __shared__ float gc[64], ec[64], er[64], bb[64];

  const int bh  = blockIdx.x;
  const int b   = bh >> 5, h = bh & 31, hk = h >> 1;
  const int tid = threadIdx.x;
  const int lane = tid & 31, wave = tid >> 5;
  const int rm = lane & 15, kg = lane >> 4, rh = kg * 8;
  const v8f vzero = {0.f,0.f,0.f,0.f,0.f,0.f,0.f,0.f};
  const ushort8 z8 = {0,0,0,0,0,0,0,0};

  for (int e = tid; e < 128 * 128; e += 256) Sf[e] = 0.f;
  for (int e = tid; e < (128 * 136) / 8; e += 256) *(ushort8*)(Sh + e * 8) = z8;
  __syncthreads();

  for (int cc = 0; cc < TSEQ / CHUNK; ++cc) {
    const int tok0 = b * TSEQ + cc * CHUNK;

    // ---------------- phase 0: stage chunk, gates, zero Delta ----------------
#pragma unroll
    for (int j = 0; j < 4; ++j) {
      int u = j * 256 + tid;
      int row = u >> 4, c8 = (u & 15) * 8;
      uint4 dq = *(const uint4*)(qn + ((size_t)(tok0 + row) * HK_ + hk) * KD_ + c8);
      uint4 dk = *(const uint4*)(kn + ((size_t)(tok0 + row) * HK_ + hk) * KD_ + c8);
      *(ushort8*)(Qh + row * 136 + c8) = *(const ushort8*)&dq;
      *(ushort8*)(Kh + row * 136 + c8) = *(const ushort8*)&dk;
    }
    for (int e = tid; e < (128 * 72) / 8; e += 256) *(ushort8*)(UhT + e * 8) = z8;
    if (tid < CHUNK) {
      gc[tid] = g[(size_t)(tok0 + tid) * HV_ + h];
      bb[tid] = beta[(size_t)(tok0 + tid) * HV_ + h];
    }
    __syncthreads();
    if (tid == 0) {
      float r = 0.f;
      for (int i = 0; i < CHUNK; ++i) { r += gc[i]; gc[i] = r; }
    }
    __syncthreads();
    if (tid < CHUNK) {
      ec[tid] = __expf(gc[tid]);
      er[tid] = __expf(gc[CHUNK - 1] - gc[tid]);
    }
    __syncthreads();

    // ---------------- phase 1: A = -tril(KK^T o decay,-1), attn ----------------
    for (int job = wave; job < 32; job += 8) {
      const int which = job >> 4;                 // 0: A, 1: attn
      const int mt = (job >> 2) & 3, nt = job & 3;
      const unsigned short* Ab = which ? Qh : Kh;
      v8f acc = vzero;
#pragma unroll
      for (int ks = 0; ks < 4; ++ks) {
        FragU fa = ld_afrag(Ab, 136, mt * 16 + rm, ks * 32, kg);
        FragU fb = ld_bfrag(Kh, 136, nt * 16 + rm, ks * 32, kg);
        WMMA_BF16(acc, fa, fb);
      }
#pragma unroll
      for (int i = 0; i < 8; ++i) {
        const int ri = mt * 16 + rh + i, cj = nt * 16 + rm;
        const float d = __expf(gc[ri] - gc[cj]);
        if (which) Th[ri * 72 + cj] = f32_to_bf16bits(ri >= cj ?  acc[i] * d : 0.f);
        else       Ah[ri * 72 + cj] = f32_to_bf16bits(ri >  cj ? -acc[i] * d : 0.f);
      }
    }
    __syncthreads();

    // ---------------- phase 2: rhs = V - ec o (K S0); o1 = ec o (Q S0) -------
    for (int job = wave; job < 64; job += 8) {
      const int which = job >> 5;                 // 0: rhs, 1: o1
      const int mt = (job >> 3) & 3, nt = job & 7;
      const unsigned short* Ab = which ? Qh : Kh;
      v8f acc = vzero;
#pragma unroll
      for (int ks = 0; ks < 4; ++ks) {
        FragU fa = ld_afrag(Ab, 136, mt * 16 + rm, ks * 32, kg);
        FragU fb = ld_bfrag(Sh, 136, nt * 16 + rm, ks * 32, kg);
        WMMA_BF16(acc, fa, fb);
      }
#pragma unroll
      for (int i = 0; i < 8; ++i) {
        const int ri = mt * 16 + rh + i, cj = nt * 16 + rm;
        if (which) {
          Of[ri * 128 + cj] = ec[ri] * acc[i];
        } else {
          float vv = bf16bits_to_f32(vn[((size_t)(tok0 + ri) * HV_ + h) * VD_ + cj]);
          Rf[ri * 128 + cj] = vv - ec[ri] * acc[i];
        }
      }
    }
    __syncthreads();

    // ---------------- phase 3: blocked forward substitution for Delta --------
    for (int r = 0; r < 4; ++r) {
      {
        const int nt = wave;                      // 8 V-tiles
        v8f acc;
#pragma unroll
        for (int i = 0; i < 8; ++i)
          acc[i] = Rf[(r * 16 + rh + i) * 128 + nt * 16 + rm];
#pragma unroll
        for (int ks = 0; ks < 2; ++ks) {          // contraction over all 64 j
          FragU fa = ld_afrag(Ah, 72, r * 16 + rm, ks * 32, kg);
          FragU fb = ld_bfrag(UhT, 72, nt * 16 + rm, ks * 32, kg);
          WMMA_BF16(acc, fa, fb);                 // Ah is -A => R - A.Delta
        }
#pragma unroll
        for (int i = 0; i < 8; ++i)
          Yf[(rh + i) * 128 + nt * 16 + rm] = acc[i];
      }
      __syncthreads();
      if (tid < 128) {                            // lane-private per V-column
        const int v = tid;
        float dl[16];
#pragma unroll
        for (int i = 0; i < 16; ++i) {
          const int gi = r * 16 + i;
          float s = Yf[i * 128 + v];
#pragma unroll
          for (int j = 0; j < i; ++j)
            s += bf16bits_to_f32(Ah[gi * 72 + r * 16 + j]) * dl[j];
          s *= bb[gi];
          dl[i] = s;
          UhT[v * 72 + gi] = f32_to_bf16bits(s);
        }
      }
      __syncthreads();
    }

    // ---------------- phase 4: o = o1 + attn . Delta -> global ----------------
    for (int job = wave; job < 32; job += 8) {
      const int mt = job >> 3, nt = job & 7;
      v8f acc;
#pragma unroll
      for (int i = 0; i < 8; ++i)
        acc[i] = Of[(mt * 16 + rh + i) * 128 + nt * 16 + rm];
#pragma unroll
      for (int ks = 0; ks < 2; ++ks) {
        FragU fa = ld_afrag(Th, 72, mt * 16 + rm, ks * 32, kg);
        FragU fb = ld_bfrag(UhT, 72, nt * 16 + rm, ks * 32, kg);
        WMMA_BF16(acc, fa, fb);
      }
#pragma unroll
      for (int i = 0; i < 8; ++i)
        o[((size_t)(tok0 + mt * 16 + rh + i) * HV_ + h) * VD_ + nt * 16 + rm] = acc[i];
    }

    // ---------------- phase 5: Kbar^T[k][j] = er_j * K[j][k] -----------------
    for (int e = tid; e < 128 * 64; e += 256) {
      const int kk = e >> 6, j = e & 63;
      Kt[kk * 72 + j] = f32_to_bf16bits(er[j] * bf16bits_to_f32(Kh[j * 136 + kk]));
    }
    __syncthreads();

    // ---------------- phase 6: S^T = gamma S^T + Delta^T . Kbar --------------
    const float gamma = ec[CHUNK - 1];
    for (int job = wave; job < 64; job += 8) {
      const int mt = job >> 3, nt = job & 7;      // mt: v-tile, nt: k-tile
      v8f acc;
#pragma unroll
      for (int i = 0; i < 8; ++i)
        acc[i] = gamma * Sf[(mt * 16 + rh + i) * 128 + nt * 16 + rm];
#pragma unroll
      for (int ks = 0; ks < 2; ++ks) {
        FragU fa = ld_afrag(UhT, 72, mt * 16 + rm, ks * 32, kg);
        FragU fb = ld_bfrag(Kt, 72, nt * 16 + rm, ks * 32, kg);
        WMMA_BF16(acc, fa, fb);
      }
#pragma unroll
      for (int i = 0; i < 8; ++i) {
        const int vv = mt * 16 + rh + i, kk = nt * 16 + rm;
        Sf[vv * 128 + kk] = acc[i];
        Sh[vv * 136 + kk] = f32_to_bf16bits(acc[i]);
      }
    }
    __syncthreads();
  }
}

// ---------------------------------------------------------------------------
// Gated RMSNorm: x = o*silu(z); x *= rsqrt(mean(x^2)+eps)*(1+w). bf16 out.
// ---------------------------------------------------------------------------
__global__ __launch_bounds__(256)
void norm_gate_kernel(const float* __restrict__ o,
                      const unsigned short* __restrict__ qkvz,
                      const float* __restrict__ norm_w,
                      unsigned short* __restrict__ xout)
{
  const int tok  = blockIdx.x;
  const int tid  = threadIdx.x, lane = tid & 31, wave = tid >> 5;
#pragma unroll
  for (int hh = 0; hh < 4; ++hh) {
    const int h = wave * 4 + hh;
    float xv[4], ss = 0.f;
#pragma unroll
    for (int j = 0; j < 4; ++j) {
      const int i = lane + j * 32;
      float ov = o[(size_t)tok * 4096 + h * VD_ + i];
      float z  = bf16bits_to_f32(qkvz[(size_t)tok * QKVZ_N + 8192 + h * VD_ + i]);
      float x  = ov * silu_f(z);
      xv[j] = x; ss += x * x;
    }
#pragma unroll
    for (int m = 16; m >= 1; m >>= 1) ss += __shfl_xor(ss, m, 32);
    const float rs = rsqrtf(ss * (1.f / 128.f) + 1e-6f);
#pragma unroll
    for (int j = 0; j < 4; ++j) {
      const int i = lane + j * 32;
      xout[(size_t)tok * 4096 + h * VD_ + i] = f32_to_bf16bits(xv[j] * rs * (1.f + norm_w[i]));
    }
  }
}

// ---------------------------------------------------------------------------
// TDM probe (compiled, NOT launched): tensor_load_to_lds + s_wait_tensorcnt +
// cluster barrier/id. 6-arg builtin form on this toolchain.
// ---------------------------------------------------------------------------
__global__ void tdm_probe_kernel(const float* __restrict__ src, float* __restrict__ dst)
{
  __shared__ float lbuf[2048];
  const unsigned long long ga = (unsigned long long)src;

  u32x4 g0;
  g0[0] = 1u;
  g0[1] = 0u;
  g0[2] = (unsigned)(ga & 0xffffffffu);
  g0[3] = (unsigned)((ga >> 32) & 0x01ffffffu) | (2u << 30);

  i32x8 g1;
  g1[0] = (2 << 16);
  g1[1] = (int)((512u & 0xffffu) << 16);
  g1[2] = (int)((512u >> 16) | (64u << 16));
  g1[3] = (int)(32u << 16);
  g1[4] = 16;
  g1[5] = 512;
  g1[6] = 0;
  g1[7] = 0;
  i32x4 gz4 = {0, 0, 0, 0};
  i32x8 gz8 = {0, 0, 0, 0, 0, 0, 0, 0};

  __builtin_amdgcn_tensor_load_to_lds(g0, g1, gz4, gz4, gz8, 0);
  __builtin_amdgcn_s_wait_tensorcnt(0);
  __builtin_amdgcn_s_cluster_barrier();
  const int cid = __builtin_amdgcn_cluster_id_x();
  dst[threadIdx.x] = lbuf[threadIdx.x] + (float)cid;
}

// ---------------------------------------------------------------------------
// Workspace layout (bytes)
// ---------------------------------------------------------------------------
static constexpr size_t OFF_QKVZ = 0;
static constexpr size_t SZ_QKVZ  = (size_t)NTOK * QKVZ_N * 2;
static constexpr size_t OFF_BA   = OFF_QKVZ + SZ_QKVZ;
static constexpr size_t SZ_BA    = (size_t)NTOK * 64 * 4;
static constexpr size_t OFF_QN   = OFF_BA + SZ_BA;
static constexpr size_t SZ_QN    = (size_t)NTOK * 2048 * 2;
static constexpr size_t OFF_KN   = OFF_QN + SZ_QN;
static constexpr size_t OFF_VN   = OFF_KN + SZ_QN;
static constexpr size_t SZ_VN    = (size_t)NTOK * 4096 * 2;
static constexpr size_t OFF_G    = OFF_VN + SZ_VN;
static constexpr size_t SZ_G     = (size_t)NTOK * HV_ * 4;
static constexpr size_t OFF_BETA = OFF_G + SZ_G;
static constexpr size_t OFF_O    = OFF_BETA + SZ_G;
static constexpr size_t SZ_O     = (size_t)NTOK * 4096 * 4;
static constexpr size_t OFF_X    = OFF_O + SZ_O;

// ---------------------------------------------------------------------------
extern "C" void kernel_launch(void* const* d_in, const int* in_sizes, int n_in,
                              void* d_out, int out_size, void* d_ws, size_t ws_size,
                              hipStream_t stream)
{
  (void)in_sizes; (void)n_in; (void)out_size; (void)ws_size;

  const float* hidden  = (const float*)d_in[0];
  const float* W_qkvz  = (const float*)d_in[1];
  const float* W_ba    = (const float*)d_in[2];
  const float* conv_w  = (const float*)d_in[3];
  const float* A_log   = (const float*)d_in[4];
  const float* dt_bias = (const float*)d_in[5];
  const float* norm_w  = (const float*)d_in[6];
  const float* W_out   = (const float*)d_in[7];

  char* ws = (char*)d_ws;
  unsigned short* qkvz = (unsigned short*)(ws + OFF_QKVZ);
  float*          ba   = (float*)(ws + OFF_BA);
  unsigned short* qn   = (unsigned short*)(ws + OFF_QN);
  unsigned short* kn   = (unsigned short*)(ws + OFF_KN);
  unsigned short* vn   = (unsigned short*)(ws + OFF_VN);
  float*          gbuf = (float*)(ws + OFF_G);
  float*          bbuf = (float*)(ws + OFF_BETA);
  float*          obuf = (float*)(ws + OFF_O);
  unsigned short* xbuf = (unsigned short*)(ws + OFF_X);

  const dim3 blk(256);

  gemm_wmma<false, true><<<dim3(QKVZ_N / 128, NTOK / 128), blk, 0, stream>>>(
      hidden, W_qkvz, qkvz, NTOK, QKVZ_N, DMODEL);

  gemm_wmma<false, false><<<dim3(1, NTOK / 128), blk, 0, stream>>>(
      hidden, W_ba, ba, NTOK, 64, DMODEL);

  conv_gate_kernel<<<dim3(NTOK), blk, 0, stream>>>(
      qkvz, ba, conv_w, A_log, dt_bias, qn, kn, vn, gbuf, bbuf);

  delta_chunk_kernel<<<dim3(2 * HV_), blk, 0, stream>>>(qn, kn, vn, gbuf, bbuf, obuf);

  norm_gate_kernel<<<dim3(NTOK), blk, 0, stream>>>(obuf, qkvz, norm_w, xbuf);

  gemm_wmma<true, false><<<dim3(DMODEL / 128, NTOK / 128), blk, 0, stream>>>(
      xbuf, W_out, d_out, NTOK, DMODEL, 4096);
}
